// MultiHeadAttentionBlock_3607772529240
// MI455X (gfx1250) — matrix operand order repaired and checked by reading, not compiled, not run on hardware
//
#include <hip/hip_runtime.h>
#include <hip/hip_bf16.h>

// ---------------------------------------------------------------------------
// MI455X (gfx1250) multi-head attention block, bf16 WMMA (f32 accumulate),
// async global->LDS staging (ASYNCcnt) for the GEMMs.
// ---------------------------------------------------------------------------

typedef __attribute__((ext_vector_type(16))) __bf16 bf16x16;
typedef __attribute__((ext_vector_type(8)))  __bf16 bf16x8;
typedef __attribute__((ext_vector_type(8)))  float  f32x8;

#define DEV static __device__ __forceinline__

constexpr int S_LEN  = 2048;
constexpr int DMODEL = 768;
constexpr int NHEAD  = 12;
constexpr int DK     = 64;
constexpr int BATCH  = 4;
constexpr int MROWS  = BATCH * S_LEN;   // 8192

constexpr int BM = 128, BN = 128, BK = 32;
constexpr int LDSS = 40;                // padded LDS row stride (bank spread)

DEV bf16x16 combine8(bf16x8 lo, bf16x8 hi) {
  bf16x16 r;
#pragma unroll
  for (int i = 0; i < 8; ++i) { r[i] = lo[i]; r[i + 8] = hi[i]; }
  return r;
}

DEV f32x8 wmma_bf16(bf16x16 a, bf16x16 b, f32x8 c) {
  // D = A(16x32 bf16) * B(32x16 bf16) + C(16x16 f32)
  return __builtin_amdgcn_wmma_f32_16x16x32_bf16(
      /*neg_a=*/false, a, /*neg_b=*/false, b,
      /*c_mod=*/(short)0, c, /*reuse_a=*/false, /*reuse_b=*/false);
}

// ---- DPP16 butterfly reductions within each 16-lane group (pure VALU) ----
template <int CTRL>
DEV float dpp_bcast(float x) {
  int xi = __float_as_int(x);
  int r = __builtin_amdgcn_update_dpp(xi, xi, CTRL, 0xF, 0xF, true);
  return __int_as_float(r);
}
DEV float red_max16(float x) {
  x = fmaxf(x, dpp_bcast<0xB1>(x));    // quad_perm[1,0,3,2]
  x = fmaxf(x, dpp_bcast<0x4E>(x));    // quad_perm[2,3,0,1]
  x = fmaxf(x, dpp_bcast<0x141>(x));   // row_half_mirror
  x = fmaxf(x, dpp_bcast<0x140>(x));   // row_mirror
  return x;
}
DEV float red_sum16(float x) {
  x += dpp_bcast<0xB1>(x);
  x += dpp_bcast<0x4E>(x);
  x += dpp_bcast<0x141>(x);
  x += dpp_bcast<0x140>(x);
  return x;
}

// ---- async global->LDS copy of 32 contiguous bytes (2 x B128) ----
// ISA 15.18: each byte goes from global VADDR+offset to LDS VDST+offset,
// i.e. the immediate offset applies to BOTH sides. Tracked by ASYNCcnt.
DEV void async_copy32(const __bf16* gsrc, __bf16* ldst) {
  unsigned laddr = (unsigned)(size_t)ldst;   // generic low 32 = LDS offset
  asm volatile(
      "global_load_async_to_lds_b128 %0, %1, off\n\t"
      "global_load_async_to_lds_b128 %0, %1, off offset:16"
      :: "v"(laddr), "v"(gsrc) : "memory");
}
DEV void wait_async0() {
  asm volatile("s_wait_asynccnt 0" ::: "memory");
}

// ---------------------------------------------------------------------------
// fp32 -> bf16 conversion (8 elements / thread, vectorized)
// ---------------------------------------------------------------------------
__global__ void cvt_f32_bf16x8(const float* __restrict__ s,
                               __bf16* __restrict__ d, int n8) {
  int i = blockIdx.x * blockDim.x + threadIdx.x;
  if (i < n8) {
    float4 f0 = ((const float4*)s)[i * 2 + 0];
    float4 f1 = ((const float4*)s)[i * 2 + 1];
    bf16x8 o;
    o[0] = (__bf16)f0.x; o[1] = (__bf16)f0.y;
    o[2] = (__bf16)f0.z; o[3] = (__bf16)f0.w;
    o[4] = (__bf16)f1.x; o[5] = (__bf16)f1.y;
    o[6] = (__bf16)f1.z; o[7] = (__bf16)f1.w;
    ((bf16x8*)d)[i] = o;
  }
}

// ---------------------------------------------------------------------------
// GEMM: Y[row][col] = sum_d A[row][d] * W[col][d]      (i.e. Y = A @ W^T)
// Double-buffered LDS fed by async global->LDS copies: wait+barrier at loop
// top, issue next tile's copies, then 8 WMMAs -- copies fly under compute.
// OUT_MODE 0: bf16 [B,H,S,DK]   1: bf16 [B,H,DK,S]   2: f32 [MROWS,DMODEL]
// ---------------------------------------------------------------------------
template <int OUT_MODE>
__global__ __launch_bounds__(256) void gemm_wmma_kernel(
    const __bf16* __restrict__ A, const __bf16* __restrict__ W,
    void* __restrict__ Dst) {
  __shared__ __bf16 Alds[2][BM * LDSS];
  __shared__ __bf16 Blds[2][BN * LDSS];

  const int tid  = threadIdx.x;
  const int lane = tid & 31;
  const int wave = tid >> 5;
  const int g    = lane >> 4;     // half-wave group (K interleave select)
  const int ln   = lane & 15;
  const int blockM = blockIdx.y * BM;
  const int blockN = blockIdx.x * BN;
  const int waveM  = (wave & 3) * 32;   // 4 waves along M
  const int waveN  = (wave >> 2) * 64;  // 2 waves along N

  const f32x8 zero = {0.f, 0.f, 0.f, 0.f, 0.f, 0.f, 0.f, 0.f};
  f32x8 acc[2][4];
#pragma unroll
  for (int i = 0; i < 2; ++i)
#pragma unroll
    for (int j = 0; j < 4; ++j) acc[i][j] = zero;

  const int srow = tid >> 1;          // 0..127
  const int scol = (tid & 1) * 16;    // 0 or 16

  auto async_stage = [&](int buf, int k0) {
    async_copy32(A + (size_t)(blockM + srow) * DMODEL + k0 + scol,
                 &Alds[buf][srow * LDSS + scol]);
    async_copy32(W + (size_t)(blockN + srow) * DMODEL + k0 + scol,
                 &Blds[buf][srow * LDSS + scol]);
  };

  constexpr int NT = DMODEL / BK;   // 24
  async_stage(0, 0);
  for (int kt = 0; kt < NT; ++kt) {
    const int buf = kt & 1;
    wait_async0();          // my copies for tile kt have landed in LDS
    __syncthreads();        // everyone's copies have landed
    if (kt + 1 < NT) async_stage(buf ^ 1, (kt + 1) * BK);  // fly under WMMAs

    bf16x16 afrag[2], bfrag[4];
#pragma unroll
    for (int i = 0; i < 2; ++i) {
      const __bf16* p = &Alds[buf][(waveM + i * 16 + ln) * LDSS + g * 8];
      afrag[i] = combine8(*(const bf16x8*)p, *(const bf16x8*)(p + 16));
    }
#pragma unroll
    for (int j = 0; j < 4; ++j) {
      const __bf16* p = &Blds[buf][(waveN + j * 16 + ln) * LDSS + g * 8];
      bfrag[j] = combine8(*(const bf16x8*)p, *(const bf16x8*)(p + 16));
    }
#pragma unroll
    for (int i = 0; i < 2; ++i)
#pragma unroll
      for (int j = 0; j < 4; ++j)
        acc[i][j] = wmma_bf16(afrag[i], bfrag[j], acc[i][j]);
  }

  // ---- epilogue: C/D layout -> global ----
#pragma unroll
  for (int i = 0; i < 2; ++i)
#pragma unroll
    for (int j = 0; j < 4; ++j)
#pragma unroll
      for (int r = 0; r < 8; ++r) {
        int row = blockM + waveM + i * 16 + g * 8 + r;
        int col = blockN + waveN + j * 16 + ln;
        float val = acc[i][j][r];
        if constexpr (OUT_MODE == 0) {
          int b = row >> 11, qi = row & (S_LEN - 1);
          int h = col >> 6, dd = col & 63;
          ((__bf16*)Dst)[(((size_t)(b * NHEAD + h)) * S_LEN + qi) * DK + dd] =
              (__bf16)val;
        } else if constexpr (OUT_MODE == 1) {
          int b = row >> 11, qi = row & (S_LEN - 1);
          int h = col >> 6, dd = col & 63;
          ((__bf16*)Dst)[(((size_t)(b * NHEAD + h)) * DK + dd) * S_LEN + qi] =
              (__bf16)val;
        } else {
          ((float*)Dst)[(size_t)row * DMODEL + col] = val;
        }
      }
}

// ---------------------------------------------------------------------------
// Causal flash attention per (b,h). 8 waves/block, 16 query rows per wave.
// K/V fragments batch-load global->VGPR (L2-resident); V loads fly under the
// score WMMAs + softmax. Softmax row reductions use DPP butterflies. P
// bounces through a per-wave LDS tile guarded by s_wait_dscnt.
// ---------------------------------------------------------------------------
__global__ __launch_bounds__(256) void attn_kernel(
    const __bf16* __restrict__ Qh, const __bf16* __restrict__ Kh,
    const __bf16* __restrict__ Vt, __bf16* __restrict__ Ctx) {
  __shared__ __bf16 Plds[8][16 * LDSS];

  const int tid = threadIdx.x, lane = tid & 31, wave = tid >> 5;
  const int g = lane >> 4, ln = lane & 15;
  const int bh = blockIdx.y;               // 0..47
  const int b = bh / NHEAD, h = bh % NHEAD;
  const int q0 = blockIdx.x * 128 + wave * 16;   // this wave's query base

  const __bf16* Qp = Qh + (size_t)bh * S_LEN * DK;
  const __bf16* Kp = Kh + (size_t)bh * S_LEN * DK;
  const __bf16* Vp = Vt + (size_t)bh * DK * S_LEN;

  bf16x16 aq[2];
#pragma unroll
  for (int ks = 0; ks < 2; ++ks) {
    const __bf16* p = Qp + (size_t)(q0 + ln) * DK + ks * 32 + g * 8;
    aq[ks] = combine8(*(const bf16x8*)p, *(const bf16x8*)(p + 16));
  }

  float mrow[8], lrow[8];
#pragma unroll
  for (int r = 0; r < 8; ++r) { mrow[r] = -1e30f; lrow[r] = 0.f; }
  const f32x8 zero = {0.f, 0.f, 0.f, 0.f, 0.f, 0.f, 0.f, 0.f};
  f32x8 oacc[4];
#pragma unroll
  for (int t = 0; t < 4; ++t) oacc[t] = zero;

  __bf16* P = &Plds[wave][0];
  const float scale = 0.125f;   // 1/sqrt(64)
  const int qrb = q0 + g * 8;

  for (int j0 = 0; j0 < q0 + 16; j0 += 32) {
    if (j0 + 32 < q0 + 16) {
      int pr = j0 + 32 + lane;
      if (pr > S_LEN - 1) pr = S_LEN - 1;
      __builtin_prefetch(Kp + (size_t)pr * DK, 0, 0);
    }

    // ---- batch-load K fragments (scores) and V fragments (PV) ----
    bf16x16 bk[2][2];
#pragma unroll
    for (int nt = 0; nt < 2; ++nt) {
      const __bf16* kp = Kp + (size_t)(j0 + nt * 16 + ln) * DK;
#pragma unroll
      for (int ks = 0; ks < 2; ++ks) {
        const __bf16* p = kp + ks * 32 + g * 8;
        bk[nt][ks] = combine8(*(const bf16x8*)p, *(const bf16x8*)(p + 16));
      }
    }
    bf16x16 bv[4];
#pragma unroll
    for (int t = 0; t < 4; ++t) {
      const __bf16* vp = Vp + (size_t)(t * 16 + ln) * S_LEN + j0 + g * 8;
      bv[t] = combine8(*(const bf16x8*)vp, *(const bf16x8*)(vp + 16));
    }

    // ---- scores ----
    f32x8 sacc[2] = {zero, zero};
#pragma unroll
    for (int nt = 0; nt < 2; ++nt)
#pragma unroll
      for (int ks = 0; ks < 2; ++ks)
        sacc[nt] = wmma_bf16(aq[ks], bk[nt][ks], sacc[nt]);

    // ---- causal mask + online softmax (DPP reductions) ----
    float p0s[8], p1s[8], corr[8];
#pragma unroll
    for (int r = 0; r < 8; ++r) {
      int qr = qrb + r;
      int c0 = j0 + ln, c1 = j0 + 16 + ln;
      float s0 = (c0 <= qr) ? sacc[0][r] * scale : -1e30f;
      float s1 = (c1 <= qr) ? sacc[1][r] * scale : -1e30f;
      float mx = red_max16(fmaxf(s0, s1));
      float nm = fmaxf(mrow[r], mx);
      float e0 = __expf(s0 - nm), e1 = __expf(s1 - nm);
      float sum = red_sum16(e0 + e1);
      float c = __expf(mrow[r] - nm);
      lrow[r] = lrow[r] * c + sum;
      mrow[r] = nm;
      corr[r] = c;
      p0s[r] = e0;
      p1s[r] = e1;
    }
#pragma unroll
    for (int t = 0; t < 4; ++t)
#pragma unroll
      for (int r = 0; r < 8; ++r) oacc[t][r] *= corr[r];

    // ---- bounce P through LDS: C-layout -> A-fragment layout ----
#pragma unroll
    for (int r = 0; r < 8; ++r) {
      P[(g * 8 + r) * LDSS + ln]      = (__bf16)p0s[r];
      P[(g * 8 + r) * LDSS + 16 + ln] = (__bf16)p1s[r];
    }
    asm volatile("s_wait_dscnt 0" ::: "memory");
    bf16x16 ap;
    {
      const __bf16* p = &P[ln * LDSS + g * 8];
      ap = combine8(*(const bf16x8*)p, *(const bf16x8*)(p + 16));
    }
    asm volatile("s_wait_dscnt 0" ::: "memory");

    // ---- O += P . V_tile ----
#pragma unroll
    for (int t = 0; t < 4; ++t) oacc[t] = wmma_bf16(ap, bv[t], oacc[t]);
  }

  // ---- normalize and write context [B,S,D] ----
#pragma unroll
  for (int t = 0; t < 4; ++t)
#pragma unroll
    for (int r = 0; r < 8; ++r) {
      float val = oacc[t][r] / lrow[r];
      int qi = q0 + g * 8 + r;
      int dd = t * 16 + ln;
      Ctx[((size_t)b * S_LEN + qi) * DMODEL + h * DK + dd] = (__bf16)val;
    }
}

// ---------------------------------------------------------------------------
// Host orchestration
// ---------------------------------------------------------------------------
extern "C" void kernel_launch(void* const* d_in, const int* in_sizes, int n_in,
                              void* d_out, int out_size, void* d_ws,
                              size_t ws_size, hipStream_t stream) {
  (void)in_sizes; (void)n_in; (void)out_size; (void)ws_size;
  const float* q  = (const float*)d_in[0];
  const float* k  = (const float*)d_in[1];
  const float* v  = (const float*)d_in[2];
  // d_in[3] = mask (causal, handled analytically)
  const float* Wq = (const float*)d_in[4];
  const float* Wk = (const float*)d_in[5];
  const float* Wv = (const float*)d_in[6];
  const float* Wo = (const float*)d_in[7];
  float* out = (float*)d_out;

  char* ws = (char*)d_ws;
  size_t off = 0;
  auto alloc = [&](size_t bytes) -> void* {
    void* p = ws + off;
    off += (bytes + 255) & ~size_t(255);
    return p;
  };
  const size_t wbytes = (size_t)DMODEL * DMODEL * sizeof(__bf16);
  const size_t xbytes = (size_t)MROWS * DMODEL * sizeof(__bf16);
  const size_t hbytes = (size_t)BATCH * NHEAD * S_LEN * DK * sizeof(__bf16);
  __bf16* Wq_b = (__bf16*)alloc(wbytes);
  __bf16* Wk_b = (__bf16*)alloc(wbytes);
  __bf16* Wv_b = (__bf16*)alloc(wbytes);
  __bf16* Wo_b = (__bf16*)alloc(wbytes);
  __bf16* qb   = (__bf16*)alloc(xbytes);
  __bf16* kb   = (__bf16*)alloc(xbytes);
  __bf16* vb   = (__bf16*)alloc(xbytes);
  __bf16* Qh   = (__bf16*)alloc(hbytes);
  __bf16* Kh   = (__bf16*)alloc(hbytes);
  __bf16* Vt   = (__bf16*)alloc(hbytes);
  __bf16* Ctx  = (__bf16*)alloc(xbytes);

  const int wn8 = DMODEL * DMODEL / 8;
  const int xn8 = MROWS * DMODEL / 8;
  cvt_f32_bf16x8<<<(wn8 + 255) / 256, 256, 0, stream>>>(Wq, Wq_b, wn8);
  cvt_f32_bf16x8<<<(wn8 + 255) / 256, 256, 0, stream>>>(Wk, Wk_b, wn8);
  cvt_f32_bf16x8<<<(wn8 + 255) / 256, 256, 0, stream>>>(Wv, Wv_b, wn8);
  cvt_f32_bf16x8<<<(wn8 + 255) / 256, 256, 0, stream>>>(Wo, Wo_b, wn8);
  cvt_f32_bf16x8<<<(xn8 + 255) / 256, 256, 0, stream>>>(q, qb, xn8);
  cvt_f32_bf16x8<<<(xn8 + 255) / 256, 256, 0, stream>>>(k, kb, xn8);
  cvt_f32_bf16x8<<<(xn8 + 255) / 256, 256, 0, stream>>>(v, vb, xn8);

  dim3 gg(DMODEL / BN, MROWS / BM);  // (6, 64)
  gemm_wmma_kernel<0><<<gg, 256, 0, stream>>>(qb, Wq_b, Qh);
  gemm_wmma_kernel<0><<<gg, 256, 0, stream>>>(kb, Wk_b, Kh);
  gemm_wmma_kernel<1><<<gg, 256, 0, stream>>>(vb, Wv_b, Vt);

  dim3 ga(S_LEN / 128, BATCH * NHEAD);  // (16, 48)
  attn_kernel<<<ga, 256, 0, stream>>>(Qh, Kh, Vt, Ctx);

  gemm_wmma_kernel<2><<<gg, 256, 0, stream>>>(Ctx, Wo_b, out);
}
